// ImprovedIPRMPNNModel_89876485636293
// MI455X (gfx1250) — matrix operand order repaired and checked by reading, not compiled, not run on hardware
//
#include <hip/hip_runtime.h>
#include <hip/hip_bf16.h>

// Problem constants (from reference): G=1024, N_PER_G=64, V=32, H=128,
// DIN=128, DOUT=10, E=1048576, N=65536.
#define G_     1024
#define NPG    64
#define V_     32
#define H_     128
#define DOUT_  10
#define EDGES  1048576
#define NNODES (G_ * NPG)

typedef __attribute__((ext_vector_type(16))) __bf16 v16bf;
typedef __attribute__((ext_vector_type(8)))  float  v8f;

__device__ __forceinline__ v8f wmma_bf16(v16bf a, v16bf b, v8f c) {
  // (neg_a, A, neg_b, B, c_mod, C, reuse_a, reuse_b)
  return __builtin_amdgcn_wmma_f32_16x16x32_bf16(false, a, false, b, (short)0, c,
                                                 false, false);
}

// A fragment (16x32 bf16, MxK) per ISA layout:
// lanes 0-15: row m0+lane, K = {0..7, 16..23} of this 32-wide K-step
// lanes 16-31: row m0+(lane-16), K = {8..15, 24..31}
template <typename T>
__device__ __forceinline__ v16bf load_a_frag(const T* Ar, int kk, int lane) {
  const int kbase = kk * 32 + ((lane & 16) ? 8 : 0);
  v16bf a;
#pragma unroll
  for (int i = 0; i < 8; ++i) a[i] = (__bf16)Ar[kbase + i];
#pragma unroll
  for (int i = 0; i < 8; ++i) a[8 + i] = (__bf16)Ar[kbase + 16 + i];
  return a;
}

// B fragment (32x16 bf16, KxN): lane = column n0+(lane%16), K = {0..7, 16..23}
// for lanes 0-15 and {8..15, 24..31} for lanes 16-31 (matches A).
__device__ __forceinline__ v16bf load_b_frag(const __bf16* Wl, int ldn, int kk,
                                             int n0, int lane) {
  const int kbase = kk * 32 + ((lane & 16) ? 8 : 0);
  const __bf16* Bc = Wl + n0 + (lane & 15);
  v16bf b;
#pragma unroll
  for (int i = 0; i < 8; ++i) b[i] = Bc[(kbase + i) * ldn];
#pragma unroll
  for (int i = 0; i < 8; ++i) b[8 + i] = Bc[(kbase + 16 + i) * ldn];
  return b;
}

// C[rows,128] = optional_relu(A[rows,128] @ W[128,128] + bias)
// grid.x = rows/128, block = 256 (8 waves, 16 rows each)
__global__ void __launch_bounds__(256) gemm_n128_bf16(
    const float* __restrict__ A, const float* __restrict__ W,
    const float* __restrict__ bias, float* __restrict__ C, int do_relu) {
  __shared__ __align__(32) __bf16 sW[128 * 128];
  __shared__ float sBias[128];
  const int t = threadIdx.x;
  for (int i = t; i < 128 * 128; i += 256) sW[i] = (__bf16)W[i];
  if (t < 128) sBias[t] = bias ? bias[t] : 0.0f;
  __syncthreads();

  const int wave = t >> 5, lane = t & 31;
  const long row0 = (long)blockIdx.x * 128 + wave * 16;
  const float* Ar = A + (row0 + (lane & 15)) * 128;

  v16bf afr[4];
#pragma unroll
  for (int kk = 0; kk < 4; ++kk) afr[kk] = load_a_frag(Ar, kk, lane);

  const int cl = lane & 15;
  const int rofs = (lane & 16) ? 8 : 0;
#pragma unroll
  for (int nt = 0; nt < 8; ++nt) {
    v8f acc = {0.f, 0.f, 0.f, 0.f, 0.f, 0.f, 0.f, 0.f};
#pragma unroll
    for (int kk = 0; kk < 4; ++kk)
      acc = wmma_bf16(afr[kk], load_b_frag(sW, 128, kk, nt * 16, lane), acc);
    const float bv = sBias[nt * 16 + cl];
    float* Cp = C + (row0 + rofs) * 128 + nt * 16 + cl;
#pragma unroll
    for (int r = 0; r < 8; ++r) {
      float v = acc[r] + bv;
      if (do_relu) v = fmaxf(v, 0.0f);
      Cp[(long)r * 128] = v;
    }
  }
}

__global__ void __launch_bounds__(256) deg_init(float* __restrict__ deg) {
  deg[(long)blockIdx.x * 256 + threadIdx.x] = 1.0f;  // self-loop
}

__global__ void __launch_bounds__(256) deg_count(const int* __restrict__ dst,
                                                 float* __restrict__ deg) {
  const int e = blockIdx.x * 256 + threadIdx.x;
  atomicAdd(&deg[dst[e]], 1.0f);
}

__global__ void __launch_bounds__(256) inv_sqrt(const float* __restrict__ deg,
                                                float* __restrict__ inv) {
  const int i = blockIdx.x * 256 + threadIdx.x;
  inv[i] = rsqrtf(deg[i]);  // deg >= 1 always
}

// acc[n][k] = b_gcn[k] + hw[n][k]*inv[n]^2   (self-loop message + bias)
__global__ void __launch_bounds__(256) acc_init(const float* __restrict__ hw,
                                                const float* __restrict__ inv,
                                                const float* __restrict__ bgcn,
                                                float* __restrict__ acc) {
  const long i = (long)blockIdx.x * 256 + threadIdx.x;
  const int n = (int)(i >> 7), k = (int)(i & 127);
  const float iv = inv[n];
  acc[i] = bgcn[k] + hw[i] * iv * iv;
}

// One wave per edge: float4 gather of hw[src], scaled, 4 L2 atomics into acc[dst].
// grid.x = E/8, block = 256 (8 waves)
__global__ void __launch_bounds__(256) gcn_scatter(
    const float* __restrict__ hw, const int* __restrict__ src,
    const int* __restrict__ dst, const float* __restrict__ inv,
    float* __restrict__ acc) {
  const int lane = threadIdx.x & 31;
  const int e = blockIdx.x * 8 + (threadIdx.x >> 5);
  const int s = src[e], d = dst[e];
  const float w = inv[s] * inv[d];
  const float4 hv = *reinterpret_cast<const float4*>(hw + (long)s * 128 + lane * 4);
  float* ap = acc + (long)d * 128 + lane * 4;
  atomicAdd(ap + 0, hv.x * w);
  atomicAdd(ap + 1, hv.y * w);
  atomicAdd(ap + 2, hv.z * w);
  atomicAdd(ap + 3, hv.w * w);
}

__global__ void __launch_bounds__(256) relu_copy(const float* __restrict__ a,
                                                 float* __restrict__ o) {
  const long i = (long)blockIdx.x * 256 + threadIdx.x;
  o[i] = fmaxf(a[i], 0.0f);
}

// Per-graph fused: att = aff@vemb^T/sqrt(H); ew = edge_w*(1+sigmoid(att));
// row-normalize; vn = ew^T @ h.  Both GEMMs via bf16 WMMA out of LDS.
// grid.x = G, block = 256
__global__ void __launch_bounds__(256) graph_attn_pool(
    const float* __restrict__ aff, const float* __restrict__ hfeat,
    const float* __restrict__ edge_w, const float* __restrict__ vemb,
    float* __restrict__ vn) {
  __shared__ __align__(32) __bf16 sAff[64 * 128];  // A for att (row-major)
  __shared__ __align__(32) __bf16 sVT[128 * 32];   // B for att: [k][v]
  __shared__ __align__(32) __bf16 sH[64 * 128];    // B for pool: [n][h]
  __shared__ __align__(32) __bf16 sEwT[32 * 64];   // A for pool: [v][n]
  __shared__ float sEw[64 * 32];

  const int g = blockIdx.x, t = threadIdx.x;
  const float* affg = aff + (long)g * 64 * 128;
  const float* hg = hfeat + (long)g * 64 * 128;
  for (int i = t; i < 64 * 128; i += 256) {
    sAff[i] = (__bf16)affg[i];
    sH[i] = (__bf16)hg[i];
  }
  const float* vg = vemb + (long)g * 32 * 128;
  for (int i = t; i < 32 * 128; i += 256) {
    const int v = i >> 7, k = i & 127;
    sVT[k * 32 + v] = (__bf16)vg[i];  // transpose to [k][v]
  }
  __syncthreads();

  const int wave = t >> 5, lane = t & 31;
  const int cl = lane & 15, rofs = (lane & 16) ? 8 : 0;

  {  // att[64,32] = sAff[64,128] @ sVT[128,32]; 8 tiles -> 1 per wave
    const int mt = wave >> 1, nt = wave & 1;
    const int m0 = mt * 16, n0 = nt * 16;
    const __bf16* Ar = sAff + (m0 + cl) * 128;
    v8f acc = {0.f, 0.f, 0.f, 0.f, 0.f, 0.f, 0.f, 0.f};
#pragma unroll
    for (int kk = 0; kk < 4; ++kk)
      acc = wmma_bf16(load_a_frag(Ar, kk, lane), load_b_frag(sVT, 32, kk, n0, lane), acc);
    const float sc = 0.08838834764831845f;  // 1/sqrt(128)
#pragma unroll
    for (int r = 0; r < 8; ++r) {
      const int rr = m0 + r + rofs, vv = n0 + cl;
      const float sig = 1.0f / (1.0f + __expf(-acc[r] * sc));
      sEw[rr * 32 + vv] = edge_w[((long)g * 64 + rr) * 32 + vv] * (1.0f + sig);
    }
  }
  __syncthreads();
  if (t < 64) {  // row-normalize over V and transpose to [v][n] bf16
    float s = 0.f;
#pragma unroll
    for (int v = 0; v < 32; ++v) s += sEw[t * 32 + v];
    const float is = (s == 0.f) ? 1.0f : 1.0f / s;
#pragma unroll
    for (int v = 0; v < 32; ++v) sEwT[v * 64 + t] = (__bf16)(sEw[t * 32 + v] * is);
  }
  __syncthreads();
  {  // vn[32,128] = sEwT[32,64] @ sH[64,128]; 16 tiles -> 2 per wave
    const int n0 = wave * 16;
#pragma unroll
    for (int mt = 0; mt < 2; ++mt) {
      const __bf16* Ar = sEwT + (mt * 16 + cl) * 64;
      v8f acc = {0.f, 0.f, 0.f, 0.f, 0.f, 0.f, 0.f, 0.f};
#pragma unroll
      for (int kk = 0; kk < 2; ++kk)
        acc = wmma_bf16(load_a_frag(Ar, kk, lane), load_b_frag(sH, 128, kk, n0, lane), acc);
      float* Vp = vn + ((long)g * 32 + mt * 16 + rofs) * 128 + n0 + cl;
#pragma unroll
      for (int r = 0; r < 8; ++r) Vp[(long)r * 128] = acc[r];
    }
  }
}

// gf[g][h] = mean over V of vn2[g][v][h]
__global__ void __launch_bounds__(128) mean_v(const float* __restrict__ vn2,
                                              float* __restrict__ gf) {
  const int g = blockIdx.x, hh = threadIdx.x;
  float s = 0.f;
#pragma unroll
  for (int v = 0; v < 32; ++v) s += vn2[((long)g * 32 + v) * 128 + hh];
  gf[g * 128 + hh] = s * (1.0f / 32.0f);
}

// out[g][j] = t[g,:] @ mlp_W2[:,j] + b2[j]  (DOUT=10, tiny tail)
__global__ void __launch_bounds__(32) final_out(const float* __restrict__ tf,
                                                const float* __restrict__ W2,
                                                const float* __restrict__ b2,
                                                float* __restrict__ out) {
  const int row = blockIdx.x, lane = threadIdx.x;
  if (lane >= DOUT_) return;
  float acc = b2[lane];
#pragma unroll 8
  for (int k = 0; k < 128; ++k) acc += tf[row * 128 + k] * W2[k * DOUT_ + lane];
  out[row * DOUT_ + lane] = acc;
}

extern "C" void kernel_launch(void* const* d_in, const int* in_sizes, int n_in,
                              void* d_out, int out_size, void* d_ws, size_t ws_size,
                              hipStream_t stream) {
  (void)in_sizes; (void)n_in; (void)out_size; (void)ws_size;
  // setup_inputs() order
  const float* x       = (const float*)d_in[0];
  const int*   ei      = (const int*)d_in[1];   // [2,E]: src = ei, dst = ei+E
  /* d_in[2] = batch (implicit: contiguous 64-node blocks) */
  const float* edge_w  = (const float*)d_in[3];
  const float* vemb    = (const float*)d_in[4];
  const float* W_emb   = (const float*)d_in[5];
  const float* b_emb   = (const float*)d_in[6];
  const float* W_gcn   = (const float*)d_in[7];
  const float* b_gcn   = (const float*)d_in[8];
  const float* aff_W1  = (const float*)d_in[9];
  const float* aff_b1  = (const float*)d_in[10];
  const float* aff_W2  = (const float*)d_in[11];
  const float* aff_b2  = (const float*)d_in[12];
  const float* vn_W1   = (const float*)d_in[13];
  const float* vn_b1   = (const float*)d_in[14];
  const float* vn_W2   = (const float*)d_in[15];
  const float* vn_b2   = (const float*)d_in[16];
  const float* mlp_W1  = (const float*)d_in[17];
  const float* mlp_b1  = (const float*)d_in[18];
  const float* mlp_W2  = (const float*)d_in[19];
  const float* mlp_b2  = (const float*)d_in[20];
  float* out = (float*)d_out;

  // workspace layout (floats), with buffer reuse
  const long NF = (long)NNODES * H_;           // 8M floats = 32MB
  const long GVF = (long)G_ * V_ * H_;         // 4M floats = 16MB
  float* ws  = (float*)d_ws;
  float* b0  = ws;             // h0 -> acc -> aff -> vn2
  float* b1  = b0 + NF;        // hw -> aff1
  float* b2b = b1 + NF;        // h (post-GCN features, kept)
  float* vn  = b2b + NF;       // [G,V,H]
  float* vn1 = vn + GVF;
  float* deg = vn1 + GVF;
  float* inv = deg + NNODES;
  float* gf  = inv + NNODES;
  float* tf  = gf + (long)G_ * H_;

  const int* src = ei;
  const int* dst = ei + EDGES;

  // degree / symmetric normalization
  deg_init<<<NNODES / 256, 256, 0, stream>>>(deg);
  deg_count<<<EDGES / 256, 256, 0, stream>>>(dst, deg);
  inv_sqrt<<<NNODES / 256, 256, 0, stream>>>(deg, inv);

  // h0 = x @ W_emb + b_emb
  gemm_n128_bf16<<<NNODES / 128, 256, 0, stream>>>(x, W_emb, b_emb, b0, 0);
  // hw = h0 @ W_gcn (bias added after aggregation)
  gemm_n128_bf16<<<NNODES / 128, 256, 0, stream>>>(b0, W_gcn, nullptr, b1, 0);
  // acc = b_gcn + self-loop; then edge scatter; then relu -> h
  acc_init<<<(int)(NF / 256), 256, 0, stream>>>(b1, inv, b_gcn, b0);
  gcn_scatter<<<EDGES / 8, 256, 0, stream>>>(b1, src, dst, inv, b0);
  relu_copy<<<(int)(NF / 256), 256, 0, stream>>>(b0, b2b);

  // affinity MLP
  gemm_n128_bf16<<<NNODES / 128, 256, 0, stream>>>(b2b, aff_W1, aff_b1, b1, 1);
  gemm_n128_bf16<<<NNODES / 128, 256, 0, stream>>>(b1, aff_W2, aff_b2, b0, 0);

  // per-graph attention + weighted pooling -> vn
  graph_attn_pool<<<G_, 256, 0, stream>>>(b0, b2b, edge_w, vemb, vn);

  // virtual-node MLP (rows = G*V = 32768)
  gemm_n128_bf16<<<(int)(G_ * V_ / 128), 256, 0, stream>>>(vn, vn_W1, vn_b1, vn1, 1);
  gemm_n128_bf16<<<(int)(G_ * V_ / 128), 256, 0, stream>>>(vn1, vn_W2, vn_b2, b0, 0);

  // graph feature + head
  mean_v<<<G_, 128, 0, stream>>>(b0, gf);
  gemm_n128_bf16<<<G_ / 128, 256, 0, stream>>>(gf, mlp_W1, mlp_b1, tf, 1);
  final_out<<<G_, 32, 0, stream>>>(tf, mlp_W2, mlp_b2, out);
}
